// SincNetLayer1D_79585743995029
// MI455X (gfx1250) — matrix operand order repaired and checked by reading, not compiled, not run on hardware
//
#include <hip/hip_runtime.h>
#include <hip/hip_bf16.h>

// ---------------------------------------------------------------------------
// SincNet layer for MI455X (gfx1250, wave32, WMMA).
//   out[n,t,f] = sum_k x[n,t+k] * filt[f,k],  t in [0,15750), f in [0,80), k in [0,251)
// GEMM view per batch: O(15750x80) = A(15750x256) * B(256x80), K zero-padded.
// f16 split-precision (hi+lo) with f32 WMMA accumulation:
//   x*w ~= xh*wh + xl*wh + xh*wl   (3x v_wmma_f32_16x16x32_f16 per k-chunk)
// ---------------------------------------------------------------------------

typedef __attribute__((ext_vector_type(16))) _Float16 v16h;
typedef __attribute__((ext_vector_type(8)))  float    v8f;

#define T_IN    16000
#define T_OUT   15750      // 16000 - 251 + 1
#define NBATCH  32
#define FNUM    80
#define FSIZE   251
#define KPAD    256
#define XROW    16384      // padded row stride for x (>= T_IN + KPAD, pow2)
#define PI_F    3.14159265358979323846f

// ---------------------------------------------------------------------------
// Kernel 0: split f32 signal into f16 hi/lo pair, zero-padded to XROW.
// ---------------------------------------------------------------------------
__global__ __launch_bounds__(256) void sinc_split_x(const float* __restrict__ x,
                                                    _Float16* __restrict__ xh,
                                                    _Float16* __restrict__ xl) {
  int idx = blockIdx.x * 256 + threadIdx.x;      // 0 .. 32*16384-1
  int n = idx >> 14;
  int t = idx & (XROW - 1);
  float v = (t < T_IN) ? x[n * T_IN + t] : 0.0f;
  _Float16 h = (_Float16)v;
  xh[idx] = h;
  xl[idx] = (_Float16)(v - (float)h);
}

// ---------------------------------------------------------------------------
// Kernel 1: build the 80 sinc band-pass filters, normalize, window, split.
// One block per filter. Output layout: w[f*KPAD + k] (n-major, K-padded with 0).
// ---------------------------------------------------------------------------
__global__ __launch_bounds__(256) void sinc_build_filters(const float* __restrict__ f1,
                                                          const float* __restrict__ fb,
                                                          _Float16* __restrict__ wh,
                                                          _Float16* __restrict__ wl) {
  __shared__ float red[256];
  __shared__ float maxv_s;
  const int f = blockIdx.x;
  const int i = threadIdx.x;

  const float f1a = fabsf(f1[f]) + 20.0f / 16000.0f;
  const float f2a = f1a + fabsf(fb[f]) + 40.0f / 16000.0f;

  float tap = 0.0f;
  if (i < FSIZE) {
    if (i == (FSIZE - 1) / 2) {            // center tap: sinc(0) = 1
      tap = 2.0f * (f2a - f1a);
    } else {
      // right half index j in [0,124]; left half mirrors the right half.
      int j = (i > 125) ? (i - 126) : (124 - i);
      float a1 = 2.0f * PI_F * f1a * (float)(j + 1);
      float a2 = 2.0f * PI_F * f2a * (float)(j + 1);
      float lp1 = 2.0f * f1a * (__sinf(a1) / a1);
      float lp2 = 2.0f * f2a * (__sinf(a2) / a2);
      tap = lp2 - lp1;
    }
  }

  // block-wide max (signed, matches jnp.max) for normalization
  red[i] = (i < FSIZE) ? tap : -3.4e38f;
  __syncthreads();
  for (int s = 128; s > 0; s >>= 1) {
    if (i < s) red[i] = fmaxf(red[i], red[i + s]);
    __syncthreads();
  }
  if (i == 0) maxv_s = red[0];
  __syncthreads();
  const float maxv = maxv_s;

  float v = 0.0f;
  if (i < FSIZE) {
    // window[i] = 0.54 - 0.46*cos(2*pi*(i*251/250)/251) = 0.54 - 0.46*cos(2*pi*i/250)
    float win = 0.54f - 0.46f * __cosf(2.0f * PI_F * (float)i / 250.0f);
    v = (tap / maxv) * win;
  }
  _Float16 h = (_Float16)v;
  wh[f * KPAD + i] = h;
  wl[f * KPAD + i] = (_Float16)(v - (float)h);
}

// ---------------------------------------------------------------------------
// Kernel 2: implicit-im2col GEMM conv via v_wmma_f32_16x16x32_f16.
// Block = 256 threads = 8 waves; each wave computes a 16(M) x 80(N) strip.
// ---------------------------------------------------------------------------
__global__ __launch_bounds__(256) void sinc_conv_wmma(const _Float16* __restrict__ xh,
                                                      const _Float16* __restrict__ xl,
                                                      const _Float16* __restrict__ wh,
                                                      const _Float16* __restrict__ wl,
                                                      float* __restrict__ out) {
  const int lane = threadIdx.x & 31;
  const int wave = threadIdx.x >> 5;
  const int l    = lane & 15;     // M (A/C) or N (B/C) sub-index
  const int hs   = lane >> 4;     // half-wave select

  const int batch = blockIdx.y;
  const int m0 = blockIdx.x * 128 + wave * 16;   // first output row of this wave
  if (m0 >= T_OUT) return;                        // wave-uniform: EXEC stays all-1s

  // A row for this lane: A[m][k] = x[batch][m0 + m + k], m = l (both half-waves).
  const _Float16* __restrict__ xhrow = xh + (size_t)batch * XROW + m0 + l;
  const _Float16* __restrict__ xlrow = xl + (size_t)batch * XROW + m0 + l;

  const v8f vzero = {0.f, 0.f, 0.f, 0.f, 0.f, 0.f, 0.f, 0.f};
  v8f acc[5];
#pragma unroll
  for (int j = 0; j < 5; ++j) acc[j] = vzero;

#pragma unroll
  for (int kk = 0; kk < KPAD; kk += 32) {
    // ---- A fragments (this code's contiguous K order, same as B):
    //   lanes 0-15 : K = kk+0..15
    //   lanes 16-31: K = kk+16..31
    v16h ah, al;
    const int o0 = kk + hs * 16;
    const int o1 = kk + hs * 16 + 8;
#pragma unroll
    for (int e = 0; e < 8; ++e) {
      ah[e]     = xhrow[o0 + e];
      ah[e + 8] = xhrow[o1 + e];
      al[e]     = xlrow[o0 + e];
      al[e + 8] = xlrow[o1 + e];
    }

    // ---- B fragments: column n per lane, K contiguous per half-wave.
    //   lanes 0-15 hold K=kk..kk+15, lanes 16-31 hold K=kk+16..kk+31.
    //   w layout [n][KPAD] makes these 32B-aligned v16h loads (L2-resident).
#pragma unroll
    for (int j = 0; j < 5; ++j) {
      const int n = j * 16 + l;
      const v16h bh = *(const v16h*)(wh + (size_t)n * KPAD + kk + hs * 16);
      const v16h bl = *(const v16h*)(wl + (size_t)n * KPAD + kk + hs * 16);
      // error-compensated split product: xh*wh + xl*wh + xh*wl
      acc[j] = __builtin_amdgcn_wmma_f32_16x16x32_f16(false, ah, false, bh,
                                                      (short)0, acc[j], false, false);
      acc[j] = __builtin_amdgcn_wmma_f32_16x16x32_f16(false, al, false, bh,
                                                      (short)0, acc[j], false, false);
      acc[j] = __builtin_amdgcn_wmma_f32_16x16x32_f16(false, ah, false, bl,
                                                      (short)0, acc[j], false, false);
    }
  }

  // ---- store: C/D 16x16 f32 layout: VGPR r -> M = r + 8*hs, N = l.
#pragma unroll
  for (int r = 0; r < 8; ++r) {
    const int t = m0 + r + hs * 8;
    if (t < T_OUT) {
      float* __restrict__ orow = out + ((size_t)batch * T_OUT + t) * FNUM;
#pragma unroll
      for (int j = 0; j < 5; ++j) orow[j * 16 + l] = acc[j][r];
    }
  }
}

// ---------------------------------------------------------------------------
extern "C" void kernel_launch(void* const* d_in, const int* in_sizes, int n_in,
                              void* d_out, int out_size, void* d_ws, size_t ws_size,
                              hipStream_t stream) {
  const float* x  = (const float*)d_in[0];   // (32, 16000, 1) f32
  const float* f1 = (const float*)d_in[1];   // (80,) f32
  const float* fb = (const float*)d_in[2];   // (80,) f32
  float* out = (float*)d_out;                // (32, 15750, 80) f32

  _Float16* xh = (_Float16*)d_ws;                       // 32*16384 halves = 1 MiB
  _Float16* xl = xh + (size_t)NBATCH * XROW;            // 1 MiB
  _Float16* wh = xl + (size_t)NBATCH * XROW;            // 80*256 halves = 40 KiB
  _Float16* wl = wh + (size_t)FNUM * KPAD;              // 40 KiB

  // 0) split x to f16 hi/lo with zero-padded rows
  sinc_split_x<<<(NBATCH * XROW) / 256, 256, 0, stream>>>(x, xh, xl);
  // 1) build + normalize + window + split the 80 filters
  sinc_build_filters<<<FNUM, 256, 0, stream>>>(f1, fb, wh, wl);
  // 2) WMMA conv: grid.x covers M in strips of 128 rows (8 waves x 16)
  dim3 grid((T_OUT + 127) / 128, NBATCH);
  sinc_conv_wmma<<<grid, 256, 0, stream>>>(xh, xl, wh, wl, out);
}
